// Efficient_LSTM_Pass_80195629351512
// MI455X (gfx1250) — matrix operand order repaired and checked by reading, not compiled, not run on hardware
//
#include <hip/hip_runtime.h>
#include <math.h>

typedef __attribute__((ext_vector_type(16))) _Float16 v16h;
typedef __attribute__((ext_vector_type(8)))  float    v8f;

#define D_MODEL 768
#define H 32
#define B_SZ 64
#define T_SZ 512
#define NROW (B_SZ * T_SZ)   // 32768 rows of (b,t)

__device__ __forceinline__ float gelu_exact(float v) {
    // torch nn.GELU() default: 0.5*v*(1+erf(v/sqrt(2)))
    return 0.5f * v * (1.0f + erff(v * 0.70710678118654752f));
}

// ---------------------------------------------------------------------------
// Phase 1: down = gelu(x @ W_down.T + b_down)   [32768 x 32], K = 768
// One wave per 16-row M-tile, two 16-col N-tiles, f16 WMMA w/ f32 accum.
// ---------------------------------------------------------------------------
#define K1_WPAD 776   // LDS row pad (halves) to break stride-768 bank conflicts

__global__ __launch_bounds__(256) void k_down(const float* __restrict__ x,
                                              const float* __restrict__ Wd,
                                              const float* __restrict__ bd,
                                              float* __restrict__ down) {
    __shared__ _Float16 sW[32 * K1_WPAD];   // W_down as f16, [n][k]
    const int tid = threadIdx.x;
    for (int idx = tid; idx < 32 * D_MODEL; idx += 256) {
        int r = idx / D_MODEL, c = idx - r * D_MODEL;
        sW[r * K1_WPAD + c] = (_Float16)Wd[idx];
    }
    __syncthreads();

    const int wave = tid >> 5, lane = tid & 31;
    const int m = lane & 15, g = lane >> 4;       // A: row m, K-half g
    const int tile = blockIdx.x * 8 + wave;       // 256 blocks * 8 waves = 2048 tiles
    const int rowbase = tile * 16;

    const float* xr = x + (size_t)(rowbase + m) * D_MODEL;
    v8f acc0 = {}, acc1 = {};

    for (int kb = 0; kb < D_MODEL; kb += 32) {
        // A fragment: lanes 0-15 hold K = kb+0..7 & kb+16..23; lanes 16-31 +8
        const float* p = xr + kb + g * 8;
        float4 p0 = *(const float4*)(p);
        float4 p1 = *(const float4*)(p + 4);
        float4 p2 = *(const float4*)(p + 16);
        float4 p3 = *(const float4*)(p + 20);
        v16h a;
        a[0] = (_Float16)p0.x;  a[1] = (_Float16)p0.y;  a[2] = (_Float16)p0.z;  a[3] = (_Float16)p0.w;
        a[4] = (_Float16)p1.x;  a[5] = (_Float16)p1.y;  a[6] = (_Float16)p1.z;  a[7] = (_Float16)p1.w;
        a[8] = (_Float16)p2.x;  a[9] = (_Float16)p2.y;  a[10] = (_Float16)p2.z; a[11] = (_Float16)p2.w;
        a[12] = (_Float16)p3.x; a[13] = (_Float16)p3.y; a[14] = (_Float16)p3.z; a[15] = (_Float16)p3.w;

        // B fragments from LDS: B[k][n] = W_down[n][k]; lane holds n = m (+16),
        // K = g*8 + e (e<8), 16 + g*8 + (e-8) (e>=8)
        const _Float16* w0 = &sW[m * K1_WPAD + kb + g * 8];
        const _Float16* w1 = &sW[(16 + m) * K1_WPAD + kb + g * 8];
        v16h b0, b1;
        #pragma unroll
        for (int e = 0; e < 16; ++e) { b0[e] = w0[(e & 7) + ((e >> 3) << 4)]; b1[e] = w1[(e & 7) + ((e >> 3) << 4)]; }

        acc0 = __builtin_amdgcn_wmma_f32_16x16x32_f16(false, a, false, b0, (short)0, acc0, false, false);
        acc1 = __builtin_amdgcn_wmma_f32_16x16x32_f16(false, a, false, b1, (short)0, acc1, false, false);
    }

    // C layout: VGPR j -> row (j + 8*g), col = m (+16 for tile 1)
    const float bia0 = bd[m], bia1 = bd[16 + m];
    #pragma unroll
    for (int j = 0; j < 8; ++j) {
        const size_t row = (size_t)(rowbase + j + 8 * g);
        down[row * H + m]      = gelu_exact(acc0[j] + bia0);
        down[row * H + 16 + m] = gelu_exact(acc1[j] + bia1);
    }
}

// ---------------------------------------------------------------------------
// Phase 2: LSTM over T=512 (serial), one block per batch, 1 thread per gate.
// rec = gelu(h_t) stored as f16 (A operand of phase 3).
// ---------------------------------------------------------------------------
__global__ __launch_bounds__(128) void k_lstm(const float* __restrict__ down,
                                              const float* __restrict__ Wih,
                                              const float* __restrict__ Whh,
                                              const float* __restrict__ bih,
                                              const float* __restrict__ bhh,
                                              _Float16* __restrict__ rec) {
    __shared__ float sWih[32 * 128];   // transposed [k][gate] -> broadcast-friendly
    __shared__ float sWhh[32 * 128];
    __shared__ float sGates[128];
    __shared__ float sH[32];
    __shared__ float sDown[2][32];

    const int j = threadIdx.x;   // gate index 0..127 (PyTorch order i,f,g,o)
    const int b = blockIdx.x;

    for (int k = 0; k < 32; ++k) {
        sWih[k * 128 + j] = Wih[j * 32 + k];
        sWhh[k * 128 + j] = Whh[j * 32 + k];
    }
    const float bias = bih[j] + bhh[j];
    const float* drow = down + (size_t)b * T_SZ * H;
    if (j < 32) { sH[j] = 0.0f; sDown[0][j] = drow[j]; }
    float c = 0.0f;
    __syncthreads();

    for (int t = 0; t < T_SZ; ++t) {
        // register-prefetch next down row to hide latency under the dot products
        float pref = 0.0f;
        if (j < 32 && t + 1 < T_SZ) pref = drow[(size_t)(t + 1) * H + j];

        const int buf = t & 1;
        float acc = bias;
        #pragma unroll 8
        for (int k = 0; k < 32; ++k)
            acc += sDown[buf][k] * sWih[k * 128 + j] + sH[k] * sWhh[k * 128 + j];
        sGates[j] = acc;
        __syncthreads();

        if (j < 32) {
            const float ig = 1.0f / (1.0f + expf(-sGates[j]));
            const float fg = 1.0f / (1.0f + expf(-sGates[32 + j]));
            const float gg = tanhf(sGates[64 + j]);
            const float og = 1.0f / (1.0f + expf(-sGates[96 + j]));
            c = fg * c + ig * gg;
            const float hn = og * tanhf(c);
            sH[j] = hn;
            rec[((size_t)b * T_SZ + t) * H + j] = (_Float16)gelu_exact(hn);
            sDown[1 - buf][j] = pref;
        }
        __syncthreads();
    }
}

// ---------------------------------------------------------------------------
// Phase 3: up = gelu(rec @ W_up.T + b_up); y = up + x; LayerNorm(y)*gamma+beta
// One 16-row tile per block (256 thr / 8 waves); K=32 == one WMMA per N-tile.
// ---------------------------------------------------------------------------
#define YPAD 772   // 8*772 % 64 == 32 -> g=0/g=1 C-writes hit disjoint banks

__global__ __launch_bounds__(256) void k_up(const _Float16* __restrict__ rec,
                                            const float* __restrict__ Wu,
                                            const float* __restrict__ bu,
                                            const float* __restrict__ x,
                                            const float* __restrict__ gamma,
                                            const float* __restrict__ beta,
                                            float* __restrict__ out) {
    __shared__ float sY[16 * YPAD];
    __shared__ float sMean[16], sRstd[16];

    const int tid = threadIdx.x;
    const int wave = tid >> 5, lane = tid & 31;
    const int m = lane & 15, g = lane >> 4;
    const int rowbase = blockIdx.x * 16;

    // A fragment (same for all waves): rec rows rowbase..rowbase+15, K=0..31
    const _Float16* ar = rec + (size_t)(rowbase + m) * H + g * 8;
    v16h a;
    #pragma unroll
    for (int i = 0; i < 8; ++i) { a[i] = ar[i]; a[8 + i] = ar[16 + i]; }

    #pragma unroll
    for (int nt = 0; nt < 6; ++nt) {
        const int ntile = wave * 6 + nt;            // 8 waves * 6 = 48 N-tiles = 768
        const int nn = ntile * 16 + m;              // output column this lane owns
        const float* wr = Wu + (size_t)nn * H + g * 8;  // B[k][nn] = W_up[nn][k]
        v16h bf;
        #pragma unroll
        for (int e = 0; e < 16; ++e) bf[e] = (_Float16)wr[(e & 7) + ((e >> 3) << 4)];

        v8f acc = {};
        acc = __builtin_amdgcn_wmma_f32_16x16x32_f16(false, a, false, bf, (short)0, acc, false, false);

        const float bia = bu[nn];
        #pragma unroll
        for (int jj = 0; jj < 8; ++jj)
            sY[(jj + 8 * g) * YPAD + nn] = gelu_exact(acc[jj] + bia);
    }
    __syncthreads();

    // residual: y += x (coalesced)
    for (int idx = tid; idx < 16 * D_MODEL; idx += 256) {
        const int r = idx / D_MODEL, cc = idx - r * D_MODEL;
        sY[r * YPAD + cc] += x[(size_t)(rowbase + r) * D_MODEL + cc];
    }
    __syncthreads();

    // per-row mean/var: 16 lanes per row, shfl_xor tree inside 16-lane groups
    {
        const int r = tid >> 4, i = tid & 15;
        float s = 0.0f, sq = 0.0f;
        #pragma unroll 4
        for (int jc = 0; jc < 48; ++jc) {
            const float v = sY[r * YPAD + i + 16 * jc];
            s += v; sq += v * v;
        }
        #pragma unroll
        for (int off = 1; off < 16; off <<= 1) {
            s  += __shfl_xor(s, off, 32);
            sq += __shfl_xor(sq, off, 32);
        }
        if (i == 0) {
            const float mean = s * (1.0f / 768.0f);
            const float var  = sq * (1.0f / 768.0f) - mean * mean;
            sMean[r] = mean;
            sRstd[r] = rsqrtf(var + 1e-5f);
        }
    }
    __syncthreads();

    // normalized, coalesced store
    for (int idx = tid; idx < 16 * D_MODEL; idx += 256) {
        const int r = idx / D_MODEL, cc = idx - r * D_MODEL;
        const float v = (sY[r * YPAD + cc] - sMean[r]) * sRstd[r] * gamma[cc] + beta[cc];
        out[(size_t)(rowbase + r) * D_MODEL + cc] = v;
    }
}

// ---------------------------------------------------------------------------
extern "C" void kernel_launch(void* const* d_in, const int* in_sizes, int n_in,
                              void* d_out, int out_size, void* d_ws, size_t ws_size,
                              hipStream_t stream) {
    const float* x      = (const float*)d_in[0];
    const float* W_down = (const float*)d_in[1];
    const float* b_down = (const float*)d_in[2];
    const float* W_ih   = (const float*)d_in[3];
    const float* W_hh   = (const float*)d_in[4];
    const float* b_ih   = (const float*)d_in[5];
    const float* b_hh   = (const float*)d_in[6];
    const float* W_up   = (const float*)d_in[7];
    const float* b_up   = (const float*)d_in[8];
    const float* gamma  = (const float*)d_in[9];
    const float* beta   = (const float*)d_in[10];
    float* out = (float*)d_out;

    // workspace: down f32 [32768*32] then rec f16 [32768*32]  (~6.3 MB)
    float*    down = (float*)d_ws;
    _Float16* rec  = (_Float16*)((char*)d_ws + (size_t)NROW * H * sizeof(float));

    k_down<<<NROW / (16 * 8), 256, 0, stream>>>(x, W_down, b_down, down);
    k_lstm<<<B_SZ, 128, 0, stream>>>(down, W_ih, W_hh, b_ih, b_hh, rec);
    k_up<<<NROW / 16, 256, 0, stream>>>(rec, W_up, b_up, x, gamma, beta, out);
}
